// ChamferLoss_43052752175176
// MI455X (gfx1250) — compile-verified, matrix operand order checked by reading, not run on hardware
//
#include <hip/hip_runtime.h>
#include <hip/hip_bf16.h>

typedef __bf16 bf16_t;
typedef __attribute__((ext_vector_type(8)))  __bf16 v8bf;
typedef __attribute__((ext_vector_type(16))) __bf16 v16bf;
typedef __attribute__((ext_vector_type(8)))  float  v8f;

#define NPTS 8192
#define KDIM 128
#define BT   128            // block tile (rows x cols of distance matrix)
#define NBLK (NPTS / BT)    // 64
#define INF_BITS 0x7F800000u

// ---------------------------------------------------------------------------
// Kernel 1: per-row squared norms (exact fp32), bf16 hi/lo split of X and Y
// (x = hi + lo, lo = bf16(x - hi), so hi*hi + hi*lo + lo*hi ~ fp32 product),
// and re-init global min arrays to +INF (workspace is not preserved).
// ---------------------------------------------------------------------------
__global__ __launch_bounds__(256) void chamfer_prep(
    const float* __restrict__ X, const float* __restrict__ Y,
    bf16_t* __restrict__ Xhi, bf16_t* __restrict__ Xlo,
    bf16_t* __restrict__ Yhi, bf16_t* __restrict__ Ylo,
    float* __restrict__ x2, float* __restrict__ y2,
    unsigned* __restrict__ rowmin, unsigned* __restrict__ colmin)
{
    int i = blockIdx.x * blockDim.x + threadIdx.x;
    if (i >= NPTS) return;
    const size_t base = (size_t)i * KDIM;
    float sx = 0.f, sy = 0.f;
#pragma unroll 4
    for (int k = 0; k < KDIM; ++k) {
        float xv = X[base + k];
        float yv = Y[base + k];
        bf16_t xh = (bf16_t)xv;
        bf16_t yh = (bf16_t)yv;
        Xhi[base + k] = xh;
        Yhi[base + k] = yh;
        Xlo[base + k] = (bf16_t)(xv - (float)xh);
        Ylo[base + k] = (bf16_t)(yv - (float)yh);
        sx += xv * xv;
        sy += yv * yv;
    }
    x2[i] = sx;
    y2[i] = sy;
    rowmin[i] = INF_BITS;
    colmin[i] = INF_BITS;
}

// 16-bit A/B fragment (16x32 / 32x16): lane l15 = row(A)/col(B); per lane two
// contiguous 8-element K runs at k0 + half*8 and k0 + 16 + half*8.
__device__ inline v16bf load_frag(const bf16_t* __restrict__ rowk0, int half)
{
    v8bf lo = *(const v8bf*)(rowk0 + half * 8);
    v8bf hi = *(const v8bf*)(rowk0 + 16 + half * 8);
    return __builtin_shufflevector(lo, hi, 0, 1, 2, 3, 4, 5, 6, 7,
                                   8, 9, 10, 11, 12, 13, 14, 15);
}

// ---------------------------------------------------------------------------
// Kernel 2: 128x128 distance tile per block. Dot products via bf16x3 split on
// V_WMMA_F32_16X16X32_BF16; fold row/col mins with uint atomicMin (exact for
// clamped non-negative floats).
// ---------------------------------------------------------------------------
__global__ __launch_bounds__(256) void chamfer_tile(
    const bf16_t* __restrict__ Xhi, const bf16_t* __restrict__ Xlo,
    const bf16_t* __restrict__ Yhi, const bf16_t* __restrict__ Ylo,
    const float* __restrict__ x2, const float* __restrict__ y2,
    unsigned* __restrict__ rowmin, unsigned* __restrict__ colmin)
{
    const int bx = blockIdx.x % NBLK;       // column block (y points)
    const int by = blockIdx.x / NBLK;       // row block    (x points)
    const int wave = threadIdx.x >> 5;      // 0..7
    const int lane = threadIdx.x & 31;
    const int half = lane >> 4;
    const int l15  = lane & 15;

    const int row0 = by * BT + wave * 16;   // this wave's 16 rows
    const int col0 = bx * BT;               // block's 128 cols

    __shared__ unsigned scol[BT];
    if (threadIdx.x < BT) scol[threadIdx.x] = INF_BITS;
    __syncthreads();

    const bf16_t* __restrict__ ARhi = Xhi + (size_t)(row0 + l15) * KDIM;
    const bf16_t* __restrict__ ARlo = Xlo + (size_t)(row0 + l15) * KDIM;

    v8f acc[8];
#pragma unroll
    for (int t = 0; t < 8; ++t)
        acc[t] = (v8f){0.f, 0.f, 0.f, 0.f, 0.f, 0.f, 0.f, 0.f};

    for (int k0 = 0; k0 < KDIM; k0 += 32) {
        v16bf ah = load_frag(ARhi + k0, half);
        v16bf al = load_frag(ARlo + k0, half);
#pragma unroll
        for (int t = 0; t < 8; ++t) {
            const size_t ybase = (size_t)(col0 + t * 16 + l15) * KDIM + k0;
            v16bf bh = load_frag(Yhi + ybase, half);
            v16bf bl = load_frag(Ylo + ybase, half);
            acc[t] = __builtin_amdgcn_wmma_f32_16x16x32_bf16(
                false, ah, false, bh, (short)0, acc[t], false, false);
            acc[t] = __builtin_amdgcn_wmma_f32_16x16x32_bf16(
                false, ah, false, bl, (short)0, acc[t], false, false);
            acc[t] = __builtin_amdgcn_wmma_f32_16x16x32_bf16(
                false, al, false, bh, (short)0, acc[t], false, false);
        }
    }

    // Epilogue. C/D layout: lane = col (n = l15), VGPR v = row m = v + half*8.
    float x2r[8];
#pragma unroll
    for (int v = 0; v < 8; ++v) x2r[v] = x2[row0 + half * 8 + v];

    float rmin[8];
#pragma unroll
    for (int v = 0; v < 8; ++v) rmin[v] = __builtin_huge_valf();

#pragma unroll
    for (int t = 0; t < 8; ++t) {
        const float y2c = y2[col0 + t * 16 + l15];
        float cmin = __builtin_huge_valf();
#pragma unroll
        for (int v = 0; v < 8; ++v) {
            float d = x2r[v] + y2c - 2.0f * acc[t][v];
            d = fmaxf(d, 0.0f);
            rmin[v] = fminf(rmin[v], d);
            cmin = fminf(cmin, d);
        }
        // lanes l and l^16 hold the same column -> fold halves, then LDS min.
        cmin = fminf(cmin, __shfl_xor(cmin, 16, 32));
        if (half == 0) atomicMin(&scol[t * 16 + l15], __float_as_uint(cmin));
    }

    // Row mins: reduce across the 16 lanes of each half.
#pragma unroll
    for (int v = 0; v < 8; ++v) {
        float r = rmin[v];
        r = fminf(r, __shfl_xor(r, 1, 32));
        r = fminf(r, __shfl_xor(r, 2, 32));
        r = fminf(r, __shfl_xor(r, 4, 32));
        r = fminf(r, __shfl_xor(r, 8, 32));
        if (l15 == 0)
            atomicMin(&rowmin[row0 + half * 8 + v], __float_as_uint(r));
    }

    __syncthreads();
    if (threadIdx.x < BT)
        atomicMin(&colmin[col0 + threadIdx.x], scol[threadIdx.x]);
}

// ---------------------------------------------------------------------------
// Kernel 3: loss = (sum(rowmin) + sum(colmin)) / N  (fixed order -> determ.)
// ---------------------------------------------------------------------------
__global__ __launch_bounds__(1024) void chamfer_reduce(
    const unsigned* __restrict__ rowmin, const unsigned* __restrict__ colmin,
    float* __restrict__ out)
{
    __shared__ float sdata[1024];
    float s = 0.f;
    for (int i = threadIdx.x; i < NPTS; i += 1024)
        s += __uint_as_float(rowmin[i]) + __uint_as_float(colmin[i]);
    sdata[threadIdx.x] = s;
    __syncthreads();
#pragma unroll
    for (int o = 512; o > 0; o >>= 1) {
        if (threadIdx.x < (unsigned)o) sdata[threadIdx.x] += sdata[threadIdx.x + o];
        __syncthreads();
    }
    if (threadIdx.x == 0) out[0] = sdata[0] / (float)NPTS;
}

// ---------------------------------------------------------------------------
extern "C" void kernel_launch(void* const* d_in, const int* in_sizes, int n_in,
                              void* d_out, int out_size, void* d_ws, size_t ws_size,
                              hipStream_t stream)
{
    const float* X = (const float*)d_in[0];   // [8192,128] fp32
    const float* Y = (const float*)d_in[1];   // [8192,128] fp32

    const size_t MAT = (size_t)NPTS * KDIM;   // 1M elements

    bf16_t* Xhi = (bf16_t*)d_ws;              // 2 MB each
    bf16_t* Xlo = Xhi + MAT;
    bf16_t* Yhi = Xlo + MAT;
    bf16_t* Ylo = Yhi + MAT;
    float*  x2  = (float*)(Ylo + MAT);        // 8192 f32 (8 MB offset, aligned)
    float*  y2  = x2 + NPTS;
    unsigned* rowmin = (unsigned*)(y2 + NPTS);
    unsigned* colmin = rowmin + NPTS;

    chamfer_prep<<<NPTS / 256, 256, 0, stream>>>(X, Y, Xhi, Xlo, Yhi, Ylo,
                                                 x2, y2, rowmin, colmin);
    chamfer_tile<<<NBLK * NBLK, 256, 0, stream>>>(Xhi, Xlo, Yhi, Ylo,
                                                  x2, y2, rowmin, colmin);
    chamfer_reduce<<<1, 1024, 0, stream>>>(rowmin, colmin, (float*)d_out);
}
